// LayoutLMPageSelfAttention_86079734546433
// MI455X (gfx1250) — matrix operand order repaired and checked by reading, not compiled, not run on hardware
//
#include <hip/hip_runtime.h>

// ---------------------------------------------------------------------------
// LayoutLM self-attention with relative_key_query bias, CDNA5 (gfx1250).
// All GEMMs via v_wmma_f32_16x16x32_bf16, flash-style online softmax,
// async global->LDS staging of K tiles.
// ---------------------------------------------------------------------------

constexpr int Bb = 4, Ss = 1024, Hh = 768, NH = 12, HD = 64;
constexpr int MAXPOS = 1024;
constexpr int EROWS = 2048;   // 2*MAXPOS-1 real rows + zero pad row

typedef __attribute__((ext_vector_type(16))) __bf16 v16bf;
typedef __attribute__((ext_vector_type(8)))  float  v8f;
typedef __attribute__((ext_vector_type(4)))  float  v4f;
typedef __attribute__((ext_vector_type(16))) unsigned short v16u16;
typedef __attribute__((ext_vector_type(8)))  unsigned short v8u16;
typedef __attribute__((ext_vector_type(4)))  int v4i;

// Pointer types for the async global->LDS copy builtin
// (param 0 is int4 in AS1/global, param 1 is int4 in AS3/LDS).
typedef __attribute__((address_space(1))) v4i* as1_v4i_ptr;
typedef __attribute__((address_space(3))) v4i* as3_v4i_ptr;

#if __has_builtin(__builtin_amdgcn_global_load_async_to_lds_b128) && \
    __has_builtin(__builtin_amdgcn_s_wait_asynccnt)
#define HAVE_ASYNC_LDS 1
#else
#define HAVE_ASYNC_LDS 0
#endif

// Hardware f32 -> bf16 convert (v_cvt_*bf16_f32), RNE.
static __device__ __forceinline__ unsigned short f2bf(float f) {
  __bf16 h = (__bf16)f;
  return __builtin_bit_cast(unsigned short, h);
}

// B-operand fragment: 16 contiguous bf16 (one column chunk of B = one row of
// the transposed source).  Lane c holds column c&15, K = (c>>4)*16 .. +15.
static __device__ __forceinline__ v16u16 ld_bfrag(const unsigned short* p) {
  v8u16 lo = *(const v8u16*)p;
  v8u16 hi = *(const v8u16*)(p + 8);
  v16u16 r;
#pragma unroll
  for (int i = 0; i < 8; ++i) { r[i] = lo[i]; r[i + 8] = hi[i]; }
  return r;
}

// A-operand fragment: lane L holds row L&15, K = (L>>4)*16 .. +15.
// Caller passes p = row_base + half*16.
static __device__ __forceinline__ v16u16 ld_afrag(const unsigned short* p) {
  v8u16 lo = *(const v8u16*)p;
  v8u16 hi = *(const v8u16*)(p + 8);
  v16u16 r;
#pragma unroll
  for (int i = 0; i < 8; ++i) { r[i] = lo[i]; r[i + 8] = hi[i]; }
  return r;
}

static __device__ __forceinline__ v8f wmma_bf16(v16u16 a, v16u16 b, v8f c) {
  return __builtin_amdgcn_wmma_f32_16x16x32_bf16(
      false, __builtin_bit_cast(v16bf, a),
      false, __builtin_bit_cast(v16bf, b),
      (short)0, c, false, false);
}

// Stage 32 rows x 64 bf16 (4 KB) of K into per-wave LDS.
// Async path: 8x global_load_async_to_lds_b128 (each lane moves 16B).
static __device__ __forceinline__ void stage_k_tile(
    const unsigned short* khr0, unsigned short* sK, int lane) {
  const char* g = (const char*)khr0;
  char* l = (char*)sK;
#if HAVE_ASYNC_LDS
#pragma unroll
  for (int i = 0; i < 8; ++i) {
    const int off = i * 512 + lane * 16;
    __builtin_amdgcn_global_load_async_to_lds_b128(
        (as1_v4i_ptr)(g + off), (as3_v4i_ptr)(l + off), 0, 0);
  }
  __builtin_amdgcn_s_wait_asynccnt(0);
#else
#pragma unroll
  for (int i = 0; i < 8; ++i) {
    const int off = i * 512 + lane * 16;
    *(v8u16*)(l + off) = *(const v8u16*)(g + off);
  }
#endif
}

// ---------------------------------------------------------------------------
// Prep kernels (all hardware-cvt, one-time):
//   weights -> bf16 transposed [out][in]; dist_emb -> bf16 padded;
//   hidden -> bf16; log-mask fused with attention_mask.
// ---------------------------------------------------------------------------
__global__ void prep_w_kernel(const float* __restrict__ Wq,
                              const float* __restrict__ Wk,
                              const float* __restrict__ Wv,
                              unsigned short* __restrict__ Wt) {
  const int idx = blockIdx.x * blockDim.x + threadIdx.x;
  if (idx >= 3 * Hh * Hh) return;
  const int mat = idx / (Hh * Hh);
  const int r   = idx % (Hh * Hh);
  const int n   = r / Hh;   // output col -> row of Wt
  const int k   = r % Hh;   // input dim  -> contiguous in Wt
  const float* W = (mat == 0) ? Wq : (mat == 1 ? Wk : Wv);
  Wt[idx] = f2bf(W[k * Hh + n]);
}

__global__ void prep_e_kernel(const float* __restrict__ E,
                              unsigned short* __restrict__ Eb) {
  const int idx = blockIdx.x * blockDim.x + threadIdx.x;
  if (idx >= EROWS * HD) return;
  const int row = idx >> 6;
  Eb[idx] = (row < 2 * MAXPOS - 1) ? f2bf(E[idx]) : (unsigned short)0;
}

__global__ void prep_h_kernel(const float* __restrict__ hidden,
                              unsigned short* __restrict__ hb) {
  const int idx = (blockIdx.x * blockDim.x + threadIdx.x) * 8;
  if (idx >= Bb * Ss * Hh) return;
  v4f a = *(const v4f*)(hidden + idx);
  v4f b = *(const v4f*)(hidden + idx + 4);
  v8u16 r;
#pragma unroll
  for (int i = 0; i < 4; ++i) { r[i] = f2bf(a[i]); r[4 + i] = f2bf(b[i]); }
  *(v8u16*)(hb + idx) = r;
}

__global__ void prep_lm_kernel(const float* __restrict__ rel,
                               const float* __restrict__ amask,
                               float* __restrict__ lm) {
  const int idx = blockIdx.x * blockDim.x + threadIdx.x;
  if (idx >= Bb * Ss * Ss) return;
  const int b = idx >> 20;          // / (S*S)
  const int r = idx & (Ss - 1);
  lm[idx] = __logf(fmaxf(rel[idx], 1e-5f)) + amask[b * Ss + r];
}

// ---------------------------------------------------------------------------
// QKV projection: one wave computes one 16x16 output tile, K=768 in 24 WMMAs.
// Q,K stored [B][NH][S][HD] bf16; V stored transposed [B][NH][HD][S] bf16.
// ---------------------------------------------------------------------------
__global__ __launch_bounds__(256) void qkv_kernel(
    const unsigned short* __restrict__ hb, const unsigned short* __restrict__ Wt,
    const float* __restrict__ bq, const float* __restrict__ bk,
    const float* __restrict__ bv,
    unsigned short* __restrict__ qb, unsigned short* __restrict__ kb,
    unsigned short* __restrict__ vbT) {
  const int lane = threadIdx.x & 31;
  const int half = lane >> 4;
  const int lr   = lane & 15;
  const int wave = blockIdx.x * 8 + (threadIdx.x >> 5);
  const int TM = (Bb * Ss) / 16;   // 256
  const int TN = Hh / 16;          // 48
  const int mat = wave / (TM * TN);
  const int rem = wave % (TM * TN);
  const int tm = rem / TN;
  const int tn = rem % TN;

  const float* bias = (mat == 0) ? bq : (mat == 1 ? bk : bv);
  const unsigned short* Wm = Wt + (size_t)mat * Hh * Hh;
  const unsigned short* ap = hb + (size_t)(tm * 16 + lr) * Hh + half * 16;
  const unsigned short* bp = Wm + (size_t)(tn * 16 + lr) * Hh + half * 16;

  v8f acc = {};
  for (int k = 0; k < Hh; k += 32)
    acc = wmma_bf16(ld_afrag(ap + k), ld_bfrag(bp + k), acc);

  const float bn = bias[tn * 16 + lr];
  const int col = tn * 16 + lr;
  const int nh = col >> 6, hd = col & 63;

  if (mat < 2) {
    unsigned short* dst = (mat == 0) ? qb : kb;
#pragma unroll
    for (int v = 0; v < 8; ++v) {
      const int rowg = tm * 16 + v + 8 * half;      // D row m = v + 8*(c>>4)
      const int bi = rowg >> 10, s = rowg & 1023;
      dst[((size_t)(bi * NH + nh) * Ss + s) * HD + hd] = f2bf(acc[v] + bn);
    }
  } else {
    v8u16 pk;
#pragma unroll
    for (int v = 0; v < 8; ++v) pk[v] = f2bf(acc[v] + bn);
    const int rowg0 = tm * 16 + 8 * half;           // 8 consecutive s values
    const int bi = rowg0 >> 10, s0 = rowg0 & 1023;
    *(v8u16*)(vbT + ((size_t)(bi * NH + nh) * HD + hd) * Ss + s0) = pk;
  }
}

// ---------------------------------------------------------------------------
// Flash attention with relative_key_query bias.
// One wave per (b, h, 16-row l-tile).  32-column r-steps:
//   async-stage K tile to LDS; scores (2 wmma) + M1=Q*E^T (4) + M2=K*E^T (4)
//   per 16 cols; diagonal gather through per-wave LDS (region aliased with
//   the K staging buffer); joint online softmax over 32 cols; P@V (4 wmma).
// ---------------------------------------------------------------------------
__global__ __launch_bounds__(256) void attn_kernel(
    const unsigned short* __restrict__ qb, const unsigned short* __restrict__ kb,
    const unsigned short* __restrict__ vbT, const unsigned short* __restrict__ Eb,
    const float* __restrict__ lm, float* __restrict__ out) {
  __shared__ float sM[8][1024];            // per-wave 4KB: K staging, then M1/M2
  __shared__ unsigned short sP[8][512];    // per-wave: probs 16x32 bf16

  const int w    = threadIdx.x >> 5;
  const int lane = threadIdx.x & 31;
  const int half = lane >> 4;
  const int lr   = lane & 15;

  const int wave = blockIdx.x * 8 + w;
  const int b   = wave / (NH * 64);
  const int rem = wave % (NH * 64);
  const int h   = rem >> 6;
  const int l0  = (rem & 63) << 4;

  const unsigned short* qh = qb  + (size_t)((b * NH + h) * Ss) * HD;
  const unsigned short* kh = kb  + (size_t)((b * NH + h) * Ss) * HD;
  const unsigned short* vh = vbT + (size_t)((b * NH + h) * HD) * Ss;
  const float* lmp = lm + (size_t)(b * Ss + l0) * Ss;

  // Q A-fragments live for the whole row sweep (K-dim = HD = 64 -> 2 frags)
  const v16u16 aq0 = ld_afrag(qh + (l0 + lr) * HD +      half * 16);
  const v16u16 aq1 = ld_afrag(qh + (l0 + lr) * HD + 32 + half * 16);

  v8f ctx0 = {}, ctx1 = {}, ctx2 = {}, ctx3 = {};
  float mrow[8], lrow[8];
#pragma unroll
  for (int v = 0; v < 8; ++v) { mrow[v] = -1e30f; lrow[v] = 0.f; }

  float* M1 = &sM[w][0];
  float* M2 = &sM[w][512];
  unsigned short* sK = (unsigned short*)&sM[w][0];   // aliases M1/M2
  unsigned short* P  = &sP[w][0];

  for (int r0 = 0; r0 < Ss; r0 += 32) {
    if (r0 + 32 < Ss) {
      __builtin_prefetch(vh + lr * Ss + r0 + 32, 0, 1);   // global_prefetch_b8
      __builtin_prefetch(lmp + lr * Ss + r0 + 32, 0, 1);
    }

    // Async-stage K rows r0..r0+31 (read once from L2, used twice below).
    stage_k_tile(kh + (size_t)r0 * HD, sK, lane);

    // Preload ALL K fragments into registers before M1/M2 overwrite sK.
    v16u16 bkf[2][2], akf[2][2];
#pragma unroll
    for (int sub = 0; sub < 2; ++sub) {
      const unsigned short* row = sK + (sub * 16 + lr) * HD;
      bkf[sub][0] = ld_bfrag(row +      half * 16);
      bkf[sub][1] = ld_bfrag(row + 32 + half * 16);
      akf[sub][0] = ld_afrag(row +      half * 16);
      akf[sub][1] = ld_afrag(row + 32 + half * 16);
    }

    float sc[2][8];
#pragma unroll
    for (int sub = 0; sub < 2; ++sub) {
      const int rt = r0 + sub * 16;

      // S = Q * K^T
      v8f s = {};
      s = wmma_bf16(aq0, bkf[sub][0], s);
      s = wmma_bf16(aq1, bkf[sub][1], s);

      // Positional bias slice: E rows Dbase-15 .. Dbase+16 (row 2047 = pad)
      const int Db = l0 - rt + (MAXPOS - 1);
      const unsigned short* e0 = Eb + (size_t)(Db - 15 + lr) * HD + half * 16;
      const unsigned short* e1 = Eb + (size_t)(Db + 1  + lr) * HD + half * 16;
      const v16u16 e00 = ld_bfrag(e0), e01 = ld_bfrag(e0 + 32);
      const v16u16 e10 = ld_bfrag(e1), e11 = ld_bfrag(e1 + 32);

      v8f m1a = {}; m1a = wmma_bf16(aq0, e00, m1a); m1a = wmma_bf16(aq1, e01, m1a);
      v8f m1b = {}; m1b = wmma_bf16(aq0, e10, m1b); m1b = wmma_bf16(aq1, e11, m1b);
      v8f m2a = {}; m2a = wmma_bf16(akf[sub][0], e00, m2a);
      m2a = wmma_bf16(akf[sub][1], e01, m2a);
      v8f m2b = {}; m2b = wmma_bf16(akf[sub][0], e10, m2b);
      m2b = wmma_bf16(akf[sub][1], e11, m2b);

      // Stash bias tiles in per-wave LDS for the diagonal gather
#pragma unroll
      for (int v = 0; v < 8; ++v) {
        const int li = v + 8 * half;
        M1[li * 32 +      lr] = m1a[v];
        M1[li * 32 + 16 + lr] = m1b[v];
        M2[li * 32 +      lr] = m2a[v];
        M2[li * 32 + 16 + lr] = m2b[v];
      }
#pragma unroll
      for (int v = 0; v < 8; ++v) {
        const int li = v + 8 * half;            // score row
        const int jj = 15 + li - lr;            // E-slice column, in [0,30]
        const float bias = M1[li * 32 + jj] + M2[lr * 32 + jj];
        sc[sub][v] = (s[v] + bias) * 0.125f + lmp[li * Ss + rt + lr];
      }
    }

    // Joint online softmax over the 32 columns (half-wave reductions:
    // xor masks 1..8 stay inside the 16-lane group owning each row)
#pragma unroll
    for (int v = 0; v < 8; ++v) {
      float t = fmaxf(sc[0][v], sc[1][v]);
#pragma unroll
      for (int m = 1; m <= 8; m <<= 1) t = fmaxf(t, __shfl_xor(t, m, 32));
      const float mn = fmaxf(mrow[v], t);
      const float scale = __expf(mrow[v] - mn);
      const float p0 = __expf(sc[0][v] - mn);
      const float p1 = __expf(sc[1][v] - mn);
      float su = p0 + p1;
#pragma unroll
      for (int m = 1; m <= 8; m <<= 1) su += __shfl_xor(su, m, 32);
      lrow[v] = lrow[v] * scale + su;
      mrow[v] = mn;
      ctx0[v] *= scale; ctx1[v] *= scale; ctx2[v] *= scale; ctx3[v] *= scale;
      const int li = v + 8 * half;
      P[li * 32 +      lr] = f2bf(p0);
      P[li * 32 + 16 + lr] = f2bf(p1);
    }

    // ctx += P(16x32) @ V(32x64)  -- V columns contiguous via vbT layout
    const v16u16 ap  = ld_afrag(P + lr * 32 + half * 16);
    const v16u16 bv0 = ld_bfrag(vh + ( 0 + lr) * Ss + r0 + half * 16);
    const v16u16 bv1 = ld_bfrag(vh + (16 + lr) * Ss + r0 + half * 16);
    const v16u16 bv2 = ld_bfrag(vh + (32 + lr) * Ss + r0 + half * 16);
    const v16u16 bv3 = ld_bfrag(vh + (48 + lr) * Ss + r0 + half * 16);
    ctx0 = wmma_bf16(ap, bv0, ctx0);
    ctx1 = wmma_bf16(ap, bv1, ctx1);
    ctx2 = wmma_bf16(ap, bv2, ctx2);
    ctx3 = wmma_bf16(ap, bv3, ctx3);
  }

#pragma unroll
  for (int v = 0; v < 8; ++v) {
    const float inv = 1.0f / lrow[v];
    const int li = v + 8 * half;
    float* op = out + (size_t)(b * Ss + l0 + li) * Hh + h * HD;
    op[ 0 + lr] = ctx0[v] * inv;
    op[16 + lr] = ctx1[v] * inv;
    op[32 + lr] = ctx2[v] * inv;
    op[48 + lr] = ctx3[v] * inv;
  }
}

// ---------------------------------------------------------------------------
extern "C" void kernel_launch(void* const* d_in, const int* in_sizes, int n_in,
                              void* d_out, int out_size, void* d_ws,
                              size_t ws_size, hipStream_t stream) {
  (void)in_sizes; (void)n_in; (void)out_size; (void)ws_size;
  const float* hidden = (const float*)d_in[0];
  const float* amask  = (const float*)d_in[1];
  const float* relm   = (const float*)d_in[2];
  const float* Wq = (const float*)d_in[3];
  const float* bq = (const float*)d_in[4];
  const float* Wk = (const float*)d_in[5];
  const float* bk = (const float*)d_in[6];
  const float* Wv = (const float*)d_in[7];
  const float* bv = (const float*)d_in[8];
  const float* dist = (const float*)d_in[9];
  float* out = (float*)d_out;

  char* ws = (char*)d_ws;
  const size_t sz_qkv = (size_t)Bb * NH * Ss * HD * 2;   // 6.29 MB each
  unsigned short* qb  = (unsigned short*)(ws);
  unsigned short* kb  = (unsigned short*)(ws + sz_qkv);
  unsigned short* vbT = (unsigned short*)(ws + 2 * sz_qkv);
  unsigned short* hb  = (unsigned short*)(ws + 3 * sz_qkv);
  unsigned short* Eb  = (unsigned short*)(ws + 4 * sz_qkv);
  unsigned short* Wt  = (unsigned short*)(ws + 4 * sz_qkv + (size_t)EROWS * HD * 2);
  float* lmb = (float*)(ws + 4 * sz_qkv + (size_t)EROWS * HD * 2 +
                        (size_t)3 * Hh * Hh * 2);

  prep_w_kernel<<<(3 * Hh * Hh + 255) / 256, 256, 0, stream>>>(Wq, Wk, Wv, Wt);
  prep_e_kernel<<<(EROWS * HD) / 256, 256, 0, stream>>>(dist, Eb);
  prep_h_kernel<<<(Bb * Ss * Hh / 8) / 256, 256, 0, stream>>>(hidden, hb);
  prep_lm_kernel<<<(Bb * Ss * Ss) / 256, 256, 0, stream>>>(relm, amask, lmb);
  qkv_kernel<<<(3 * (Bb * Ss / 16) * (Hh / 16)) / 8, 256, 0, stream>>>(
      hb, Wt, bq, bk, bv, qb, kb, vbT);
  attn_kernel<<<(Bb * NH * (Ss / 16)) / 8, 256, 0, stream>>>(
      qb, kb, vbT, Eb, lmb, out);
}
